// MambaIRUNetRetinex_10823317586418
// MI455X (gfx1250) — matrix operand order repaired and checked by reading, not compiled, not run on hardware
//
#include <hip/hip_runtime.h>
#include <hip/hip_bf16.h>
#include <math.h>

// Problem constants (reference: B,H,W,DM = 2,64,64,96; DIN,R,N,K = 192,6,16,4)
static constexpr int cB   = 2;
static constexpr int cH   = 64;
static constexpr int cW   = 64;
static constexpr int cL   = 4096;   // H*W
static constexpr int cDM  = 96;
static constexpr int cDIN = 192;
static constexpr int cK   = 4;
static constexpr int cR   = 6;
static constexpr int cN   = 16;
static constexpr int cC   = 38;     // R + 2N
static constexpr int cXZ  = 2 * cDIN; // 384

typedef __attribute__((ext_vector_type(16))) __bf16 v16bf;
typedef __attribute__((ext_vector_type(8)))  float  v8f;

__device__ __forceinline__ float silu_f(float v) { return v / (1.0f + __expf(-v)); }
__device__ __forceinline__ float softplus_f(float v) { return (v > 20.0f) ? v : log1pf(__expf(v)); }

// scan-position -> spatial (row-major h*W+w) index; bijection per direction.
// k=0: hw forward, k=1: wh forward, k=2: hw reverse, k=3: wh reverse.
__device__ __forceinline__ int dir_map(int k, int l) {
  switch (k & 3) {
    case 0:  return l;
    case 1:  return ((l & 63) << 6) | (l >> 6);            // (hh*W + ww), l = ww*H+hh
    case 2:  return (cL - 1) - l;
    default: { int lp = (cL - 1) - l; return ((lp & 63) << 6) | (lp >> 6); }
  }
}

// ---------------------------------------------------------------------------
// GEMM 1: xz(8192,384) = x(8192,96) @ in_proj_w(384,96)^T   (bf16 WMMA)
// one wave per 16x16 tile; tiles = 512 x 24 = 12288 waves
// ---------------------------------------------------------------------------
__global__ void k_gemm_inproj(const float* __restrict__ x,
                              const float* __restrict__ w,
                              float* __restrict__ xz) {
  int wid   = (blockIdx.x * blockDim.x + threadIdx.x) >> 5;
  int lane  = threadIdx.x & 31;
  int col16 = lane & 15;
  int khalf = lane >> 4;
  int mt = wid / 24;
  int nt = wid - mt * 24;
  int m0 = mt << 4, n0 = nt << 4;
  int arow = m0 + col16;
  int bcol = n0 + col16;
  v8f acc = {};
  #pragma unroll
  for (int kk0 = 0; kk0 < cDM; kk0 += 32) {
    v16bf a, bf;
    const float* ap = x + (size_t)arow * cDM + kk0;
    #pragma unroll
    for (int e = 0; e < 8; ++e) {
      a[e]     = (__bf16)ap[khalf * 16 + e];         // K = kk0 + khalf*16 + e
      a[e + 8] = (__bf16)ap[khalf * 16 + 8 + e];     // K = kk0 + khalf*16 + 8 + e
    }
    const float* bp = w + (size_t)bcol * cDM + kk0 + khalf * 16;
    #pragma unroll
    for (int e = 0; e < 16; ++e) bf[e] = (__bf16)bp[e]; // B[k][n] = w[n*96+k]
    acc = __builtin_amdgcn_wmma_f32_16x16x32_bf16(false, a, false, bf,
                                                  (short)0, acc, false, false);
  }
  float* op = xz + (size_t)n0 + col16;
  #pragma unroll
  for (int r = 0; r < 8; ++r)
    op[(size_t)(m0 + r + khalf * 8) * cXZ] = acc[r];
}

// ---------------------------------------------------------------------------
// Depthwise 3x3 SAME conv on xc half of xz, + bias, SiLU.
// Output layout (b, sidx, d)  -> xc[(b*L+sidx)*DIN + d]
// ---------------------------------------------------------------------------
__global__ void k_conv_silu(const float* __restrict__ xz,
                            const float* __restrict__ cw,
                            const float* __restrict__ cb,
                            float* __restrict__ xc) {
  int t = blockIdx.x * blockDim.x + threadIdx.x;
  if (t >= cB * cL * cDIN) return;
  int d    = t % cDIN;
  int sidx = (t / cDIN) % cL;
  int b    = t / (cDIN * cL);
  int h = sidx >> 6, wcol = sidx & 63;
  const float* base = xz + (size_t)b * cL * cXZ;
  float acc = 0.0f;
  #pragma unroll
  for (int kh = 0; kh < 3; ++kh) {
    int hh = h + kh - 1;
    if (hh < 0 || hh >= cH) continue;
    #pragma unroll
    for (int kw = 0; kw < 3; ++kw) {
      int ww = wcol + kw - 1;
      if (ww < 0 || ww >= cW) continue;
      acc += base[(size_t)(hh * cW + ww) * cXZ + d] * cw[d * 9 + kh * 3 + kw];
    }
  }
  xc[t] = silu_f(acc + cb[d]);
}

__global__ void k_zero(float* __restrict__ p, int n) {
  int t = blockIdx.x * blockDim.x + threadIdx.x;
  if (t < n) p[t] = 0.0f;
}

// ---------------------------------------------------------------------------
// GEMM 2: x_dbl[b,k](38,4096) = x_proj_w[k](38,192) @ Xs[b,k](192,4096)
// Xs gathered from xc with direction map. Waves: b(2)*k(4)*mt(3)*lt(256)=6144
// ---------------------------------------------------------------------------
__global__ void k_gemm_xdbl(const float* __restrict__ xc,
                            const float* __restrict__ xpw,
                            float* __restrict__ xdbl) {
  int wid   = (blockIdx.x * blockDim.x + threadIdx.x) >> 5;
  int lane  = threadIdx.x & 31;
  int col16 = lane & 15;
  int khalf = lane >> 4;
  int lt = wid & 255;
  int mt = (wid >> 8) % 3;
  int k  = ((wid >> 8) / 3) % cK;
  int b  = wid / (256 * 3 * cK);

  int l    = (lt << 4) + col16;
  int sidx = dir_map(k, l);
  const float* bbase = xc + ((size_t)b * cL + sidx) * cDIN;

  int  arow   = mt * 16 + col16;
  bool avalid = arow < cC;
  const float* abase = xpw + ((size_t)k * cC + (avalid ? arow : 0)) * cDIN;

  v8f acc = {};
  #pragma unroll
  for (int kk0 = 0; kk0 < cDIN; kk0 += 32) {
    v16bf a, bf;
    #pragma unroll
    for (int e = 0; e < 8; ++e) {
      float v0 = avalid ? abase[kk0 + khalf * 16 + e] : 0.0f;
      float v1 = avalid ? abase[kk0 + khalf * 16 + 8 + e] : 0.0f;
      a[e] = (__bf16)v0;
      a[e + 8] = (__bf16)v1;
    }
    const float* bp = bbase + kk0 + khalf * 16;
    #pragma unroll
    for (int e = 0; e < 16; ++e) bf[e] = (__bf16)bp[e];
    acc = __builtin_amdgcn_wmma_f32_16x16x32_bf16(false, a, false, bf,
                                                  (short)0, acc, false, false);
  }
  float* obase = xdbl + (size_t)(b * cK + k) * cC * cL + l;
  #pragma unroll
  for (int r = 0; r < 8; ++r) {
    int c = mt * 16 + r + khalf * 8;
    if (c < cC) obase[(size_t)c * cL] = acc[r];
  }
}

// ---------------------------------------------------------------------------
// Selective scan: one lane per (b,k,d). 1536 lanes; lanes of a wave share
// (b,k) (DIN=192 multiple of 32) so x_dbl row reads are wave-uniform.
// delta = softplus(dt_proj(dts) + bias) computed on the fly (R=6 FMAs).
// Scatter y + D*u atomically into output-ordered accumulator.
// ---------------------------------------------------------------------------
__global__ void k_scan(const float* __restrict__ xc,
                       const float* __restrict__ xdbl,
                       const float* __restrict__ dtw,
                       const float* __restrict__ dtb,
                       const float* __restrict__ A_logs,
                       const float* __restrict__ Ds,
                       float* __restrict__ yacc) {
  int t = blockIdx.x * blockDim.x + threadIdx.x;   // 0..1535
  int d = t % cDIN;
  int k = (t / cDIN) % cK;
  int b = t / (cDIN * cK);
  int kd = k * cDIN + d;

  const float* xd  = xdbl + (size_t)(b * cK + k) * cC * cL;
  const float* xcb = xc + (size_t)b * cL * cDIN;
  float*       yb  = yacc + (size_t)b * cL * cDIN;

  float dtwr[cR];
  #pragma unroll
  for (int r = 0; r < cR; ++r) dtwr[r] = dtw[(size_t)kd * cR + r];
  float bias = dtb[kd];
  float Acoef[cN];
  #pragma unroll
  for (int n = 0; n < cN; ++n) Acoef[n] = -__expf(A_logs[(size_t)kd * cN + n]);
  float Dv = Ds[kd];

  float hst[cN];
  #pragma unroll
  for (int n = 0; n < cN; ++n) hst[n] = 0.0f;

  for (int l = 0; l < cL; ++l) {
    if (l + 64 < cL) {
      __builtin_prefetch(xd + (size_t)cR * cL + l + 64, 0, 1);   // B rows ahead
    }
    int sidx = dir_map(k, l);                 // wave-uniform
    float u = xcb[(size_t)sidx * cDIN + d];   // coalesced across lanes (d-minor)

    float dr = bias;
    #pragma unroll
    for (int r = 0; r < cR; ++r) dr += xd[(size_t)r * cL + l] * dtwr[r];
    float delta = softplus_f(dr);
    float du = delta * u;

    float y = 0.0f;
    #pragma unroll
    for (int n = 0; n < cN; ++n) {
      float Bn = xd[(size_t)(cR + n) * cL + l];        // wave-uniform
      float Cn = xd[(size_t)(cR + cN + n) * cL + l];   // wave-uniform
      float dA = __expf(delta * Acoef[n]);
      hst[n] = dA * hst[n] + du * Bn;
      y += hst[n] * Cn;
    }
    atomicAdd(&yb[(size_t)sidx * cDIN + d], y + Dv * u);
  }
}

// ---------------------------------------------------------------------------
// LayerNorm over DIN + SiLU(z) gate. One block (192 threads) per token.
// ---------------------------------------------------------------------------
__global__ void k_ln_gate(const float* __restrict__ yacc,
                          const float* __restrict__ xz,
                          const float* __restrict__ g,
                          const float* __restrict__ be,
                          float* __restrict__ yg) {
  __shared__ float s1[cDIN];
  __shared__ float s2[cDIN];
  int row = blockIdx.x;            // b*L + l
  int tid = threadIdx.x;           // 0..191
  float v = yacc[(size_t)row * cDIN + tid];
  s1[tid] = v;
  s2[tid] = v * v;
  __syncthreads();
  for (int off = 96; off >= 3; off >>= 1) {
    if (tid < off) { s1[tid] += s1[tid + off]; s2[tid] += s2[tid + off]; }
    __syncthreads();
  }
  float mu  = (s1[0] + s1[1] + s1[2]) * (1.0f / cDIN);
  float var = (s2[0] + s2[1] + s2[2]) * (1.0f / cDIN) - mu * mu;
  float yn  = (v - mu) * rsqrtf(var + 1e-5f) * g[tid] + be[tid];
  float z   = xz[(size_t)row * cXZ + cDIN + tid];
  yg[(size_t)row * cDIN + tid] = yn * silu_f(z);
}

// ---------------------------------------------------------------------------
// GEMM 3: out(8192,96) = yg(8192,192) @ out_proj_w(96,192)^T   (bf16 WMMA)
// tiles = 512 x 6 = 3072 waves
// ---------------------------------------------------------------------------
__global__ void k_gemm_outproj(const float* __restrict__ yg,
                               const float* __restrict__ w,
                               float* __restrict__ out) {
  int wid   = (blockIdx.x * blockDim.x + threadIdx.x) >> 5;
  int lane  = threadIdx.x & 31;
  int col16 = lane & 15;
  int khalf = lane >> 4;
  int mt = wid / 6;
  int nt = wid - mt * 6;
  int m0 = mt << 4, n0 = nt << 4;
  int arow = m0 + col16;
  int bcol = n0 + col16;
  v8f acc = {};
  #pragma unroll
  for (int kk0 = 0; kk0 < cDIN; kk0 += 32) {
    v16bf a, bf;
    const float* ap = yg + (size_t)arow * cDIN + kk0;
    #pragma unroll
    for (int e = 0; e < 8; ++e) {
      a[e]     = (__bf16)ap[khalf * 16 + e];
      a[e + 8] = (__bf16)ap[khalf * 16 + 8 + e];
    }
    const float* bp = w + (size_t)bcol * cDIN + kk0 + khalf * 16;
    #pragma unroll
    for (int e = 0; e < 16; ++e) bf[e] = (__bf16)bp[e];
    acc = __builtin_amdgcn_wmma_f32_16x16x32_bf16(false, a, false, bf,
                                                  (short)0, acc, false, false);
  }
  float* op = out + (size_t)n0 + col16;
  #pragma unroll
  for (int r = 0; r < 8; ++r)
    op[(size_t)(m0 + r + khalf * 8) * cDM] = acc[r];
}

// ---------------------------------------------------------------------------
extern "C" void kernel_launch(void* const* d_in, const int* in_sizes, int n_in,
                              void* d_out, int out_size, void* d_ws, size_t ws_size,
                              hipStream_t stream) {
  const float* x          = (const float*)d_in[0];
  const float* in_proj_w  = (const float*)d_in[1];
  const float* conv_w     = (const float*)d_in[2];
  const float* conv_b     = (const float*)d_in[3];
  const float* x_proj_w   = (const float*)d_in[4];
  const float* dt_proj_w  = (const float*)d_in[5];
  const float* dt_proj_b  = (const float*)d_in[6];
  const float* A_logs     = (const float*)d_in[7];
  const float* Ds         = (const float*)d_in[8];
  const float* out_norm_g = (const float*)d_in[9];
  const float* out_norm_b = (const float*)d_in[10];
  const float* out_proj_w = (const float*)d_in[11];
  float* out = (float*)d_out;

  float* ws   = (float*)d_ws;
  const size_t tok = (size_t)cB * cL;                 // 8192
  float* xz   = ws;                                   // tok * 384
  float* xc   = xz   + tok * cXZ;                     // tok * 192 (also reused as yg)
  float* xdbl = xc   + tok * cDIN;                    // B*K*38*L
  float* yacc = xdbl + (size_t)cB * cK * cC * cL;     // tok * 192
  float* yg   = xc;   // xc dead after scan; reuse

  // 1) in_proj GEMM (WMMA bf16): 12288 waves
  k_gemm_inproj<<<1536, 256, 0, stream>>>(x, in_proj_w, xz);
  // 2) depthwise conv + SiLU
  k_conv_silu<<<(int)((tok * cDIN + 255) / 256), 256, 0, stream>>>(xz, conv_w, conv_b, xc);
  // 3) zero direction-merge accumulator
  k_zero<<<(int)((tok * cDIN + 255) / 256), 256, 0, stream>>>(yacc, (int)(tok * cDIN));
  // 4) x_dbl GEMM (WMMA bf16, direction gather): 6144 waves
  k_gemm_xdbl<<<768, 256, 0, stream>>>(xc, x_proj_w, xdbl);
  // 5) selective scan, fused delta + atomic direction merge: 48 waves
  k_scan<<<6, 256, 0, stream>>>(xc, xdbl, dt_proj_w, dt_proj_b, A_logs, Ds, yacc);
  // 6) LayerNorm + SiLU gate
  k_ln_gate<<<(int)tok, cDIN, 0, stream>>>(yacc, xz, out_norm_g, out_norm_b, yg);
  // 7) out_proj GEMM (WMMA bf16): 3072 waves
  k_gemm_outproj<<<384, 256, 0, stream>>>(yg, out_proj_w, out);

  (void)in_sizes; (void)n_in; (void)out_size; (void)ws_size;
}
